// BidirectionalMultiLayersEncoder_14181982012223
// MI455X (gfx1250) — matrix operand order repaired and checked by reading, not compiled, not run on hardware
//
#include <hip/hip_runtime.h>
#include <hip/hip_bf16.h>

typedef _Float16 half_t;
typedef __attribute__((ext_vector_type(16))) _Float16 v16h;
typedef __attribute__((ext_vector_type(8)))  _Float16 v8h;
typedef __attribute__((ext_vector_type(8)))  float    v8f;

constexpr int Bt = 32;      // batch
constexpr int Tt = 512;     // seq len
constexpr int Et = 512;     // embed
constexpr int Ht = 512;     // hidden
constexpr int G4H = 4 * Ht; // 2048 gate width

// ---------------------------------------------------------------------------
// elementwise f32 -> f16
// ---------------------------------------------------------------------------
__global__ void f32_to_f16_kernel(const float* __restrict__ in,
                                  half_t* __restrict__ out, int n) {
  int i = blockIdx.x * blockDim.x + threadIdx.x;
  if (i < n) out[i] = (half_t)in[i];
}

// ---------------------------------------------------------------------------
// transpose + convert:  in [K,N] f32 (row-major)  ->  out [N,K] f16
// so WMMA B-fragments become K-contiguous loads.
// ---------------------------------------------------------------------------
__global__ void transpose_f16_kernel(const float* __restrict__ in,
                                     half_t* __restrict__ out, int K, int N) {
  int idx = blockIdx.x * blockDim.x + threadIdx.x;
  if (idx < K * N) {
    int n = idx % N;
    int k = idx / N;
    out[(size_t)n * K + k] = (half_t)in[idx];
  }
}

// ---------------------------------------------------------------------------
// Input projection GEMM: Z[t*B+b, 2048] = X[b*T+t, K](f16) @ WT[2048,K]^T + b
// Each wave: two adjacent 16x16 N-tiles (shared A fragment) x split-K into
// two partial accumulators -> 4 independent v_wmma chains per wave.
// Output rows are permuted to the [T, B, 4H] layout the recurrence wants.
// ---------------------------------------------------------------------------
__global__ __launch_bounds__(256) void gemm_xz_kernel(
    const half_t* __restrict__ X,    // [M,K] row-major, M flat = b*T + t
    const half_t* __restrict__ WT,   // [2048,K] row-major (transposed weight)
    const float*  __restrict__ bias, // [2048]
    float*        __restrict__ Z,    // [T*B, 2048]  (row = t*B + b)
    int M, int K) {
  const int lane = threadIdx.x & 31;
  const int wave = threadIdx.x >> 5;
  const int NP = G4H / 32;              // 64 n-tile PAIRS per row of tiles
  const int pair = blockIdx.x * 8 + wave;
  const int mt = pair / NP;
  const int np = pair % NP;             // covers nt = 2*np, 2*np+1
  if (mt * 16 >= M) return;             // wave-uniform

  const int lr = lane & 15;  // row (A) / col (B,C)
  const int hs = lane >> 4;  // half-select

  v8f acc[2][2]; // [n-tile][k-phase]
#pragma unroll
  for (int j = 0; j < 2; ++j) {
    float bv = bias[(2 * np + j) * 16 + lr];
#pragma unroll
    for (int r = 0; r < 8; ++r) { acc[j][0][r] = bv; acc[j][1][r] = 0.f; }
  }

  const half_t* arow  = X  + (size_t)(mt * 16 + lr) * K;
  const half_t* brow0 = WT + (size_t)((2 * np + 0) * 16 + lr) * K;
  const half_t* brow1 = WT + (size_t)((2 * np + 1) * 16 + lr) * K;

  for (int k0 = 0; k0 < K; k0 += 64) {
    union { v16h v; v8h h[2]; } a0, a1, b00, b01, b10, b11;
    // A 16x32: lanes<16 hold K {0..7,16..23}, lanes>=16 hold K {8..15,24..31}
    a0.h[0] = *(const v8h*)(arow + k0      + hs * 8);
    a0.h[1] = *(const v8h*)(arow + k0 + 16 + hs * 8);
    a1.h[0] = *(const v8h*)(arow + k0 + 32 + hs * 8);
    a1.h[1] = *(const v8h*)(arow + k0 + 48 + hs * 8);
    // B 32x16: lanes<16 hold K {0..7,16..23}, lanes>=16 hold K {8..15,24..31}
    b00.h[0] = *(const v8h*)(brow0 + k0      + hs * 8);
    b00.h[1] = *(const v8h*)(brow0 + k0 + 16 + hs * 8);
    b10.h[0] = *(const v8h*)(brow1 + k0      + hs * 8);
    b10.h[1] = *(const v8h*)(brow1 + k0 + 16 + hs * 8);
    b01.h[0] = *(const v8h*)(brow0 + k0 + 32 + hs * 8);
    b01.h[1] = *(const v8h*)(brow0 + k0 + 48 + hs * 8);
    b11.h[0] = *(const v8h*)(brow1 + k0 + 32 + hs * 8);
    b11.h[1] = *(const v8h*)(brow1 + k0 + 48 + hs * 8);
    // 4 independent accumulation chains
    acc[0][0] = __builtin_amdgcn_wmma_f32_16x16x32_f16(false, a0.v, false, b00.v,
                                                       (short)0, acc[0][0], false, false);
    acc[1][0] = __builtin_amdgcn_wmma_f32_16x16x32_f16(false, a0.v, false, b10.v,
                                                       (short)0, acc[1][0], false, false);
    acc[0][1] = __builtin_amdgcn_wmma_f32_16x16x32_f16(false, a1.v, false, b01.v,
                                                       (short)0, acc[0][1], false, false);
    acc[1][1] = __builtin_amdgcn_wmma_f32_16x16x32_f16(false, a1.v, false, b11.v,
                                                       (short)0, acc[1][1], false, false);
  }

#pragma unroll
  for (int j = 0; j < 2; ++j) {
#pragma unroll
    for (int r = 0; r < 8; ++r) {
      const int m = mt * 16 + r + 8 * hs; // flat input row = b*T + t
      const int b = m >> 9;               // /T
      const int t = m & (Tt - 1);         // %T
      Z[((size_t)t * Bt + b) * G4H + (2 * np + j) * 16 + lr] =
          acc[j][0][r] + acc[j][1][r];
    }
  }
}

// ---------------------------------------------------------------------------
// Persistent LSTM recurrence. grid = 2 (dir), block = 1024 (32 waves).
// Wave w owns h-columns [16w,16w+16): computes i,f,g,o tiles for both M-tiles,
// so the state update is wave-local. h lives in LDS; Wh streams from L2.
// Z is [T, B, 4H]: one contiguous 256KB slab per step, prefetched one step
// ahead (global_prefetch_b8) while the current step's WMMAs run.
// ---------------------------------------------------------------------------
__global__ __launch_bounds__(1024) void lstm_rec_kernel(
    const float*  __restrict__ Zf,   // [T*B, 2048] x-projection, fwd weights
    const float*  __restrict__ Zb,   // same, bwd weights (natural t order)
    const half_t* __restrict__ WhTf, // [2048, 512] transposed recurrent W fwd
    const half_t* __restrict__ WhTb,
    float*  seq_f32,  // nullable: [B,T,1024] final output (layer 1)
    half_t* seq_f16,  // nullable: [B,T,1024] f16 feed to next layer (layer 0)
    float*  hT_out,   // [B,1024] (col offset dir*512)
    float*  cT_out) { // [B,1024]
  const int dir = blockIdx.x;
  const float*  Z   = dir ? Zb   : Zf;
  const half_t* WhT = dir ? WhTb : WhTf;

  __shared__ __align__(16) half_t hbuf[32][520]; // 520 halfs: 16B-mult stride

  const int tid  = threadIdx.x;
  const int lane = tid & 31;
  const int w    = tid >> 5; // wave id == h-column tile 0..31
  const int lr   = lane & 15;
  const int hs   = lane >> 4;

  // h0 = 0
  for (int i = tid; i < 32 * 520; i += 1024) (&hbuf[0][0])[i] = (half_t)0.f;
  __syncthreads();

  float cst[2][8]; // c state: this lane's 16 (m,row) cells
#pragma unroll
  for (int m = 0; m < 2; ++m)
#pragma unroll
    for (int r = 0; r < 8; ++r) cst[m][r] = 0.f;

  for (int t = 0; t < Tt; ++t) {
    const int tt = dir ? (Tt - 1 - t) : t;

    // ---- prefetch next step's 256KB z-slab into L2 while we compute ----
    if (t + 1 < Tt) {
      const int tn = dir ? (Tt - 2 - t) : (t + 1);
      const char* slab = (const char*)(Z + (size_t)tn * Bt * G4H);
      __builtin_prefetch(slab + (size_t)tid * 256, 0, 1);
      __builtin_prefetch(slab + (size_t)tid * 256 + 128, 0, 1);
    }

    // ---- C init: z = xz[tt] tiles (gate g, M-tile m) -------------------
    v8f acc[4][2];
    const float* zslab = Z + (size_t)tt * Bt * G4H;
#pragma unroll
    for (int g = 0; g < 4; ++g)
#pragma unroll
      for (int m = 0; m < 2; ++m) {
        const int n = g * Ht + w * 16 + lr;
#pragma unroll
        for (int r = 0; r < 8; ++r) {
          const int b = m * 16 + r + 8 * hs;
          acc[g][m][r] = zslab[(size_t)b * G4H + n];
        }
      }

    // ---- z += h_prev @ Wh  (K = 512 in chunks of 32) -------------------
    for (int k0 = 0; k0 < Ht; k0 += 32) {
      union { v16h v; v8h h[2]; } a[2];
#pragma unroll
      for (int m = 0; m < 2; ++m) {
        const half_t* ap = &hbuf[m * 16 + lr][0];
        a[m].h[0] = *(const v8h*)(ap + k0 + hs * 8);
        a[m].h[1] = *(const v8h*)(ap + k0 + 16 + hs * 8);
      }
#pragma unroll
      for (int g = 0; g < 4; ++g) {
        union { v16h v; v8h h[2]; } bf;
        const half_t* bp = WhT + (size_t)(g * Ht + w * 16 + lr) * Ht;
        bf.h[0] = *(const v8h*)(bp + k0 + hs * 8);
        bf.h[1] = *(const v8h*)(bp + k0 + 16 + hs * 8);
#pragma unroll
        for (int m = 0; m < 2; ++m)
          acc[g][m] = __builtin_amdgcn_wmma_f32_16x16x32_f16(
              false, a[m].v, false, bf.v, (short)0, acc[g][m], false, false);
      }
    }

    __syncthreads(); // everyone done READING h_prev

    // ---- gates (Keras order i,f,c,o), state update, h broadcast --------
#pragma unroll
    for (int m = 0; m < 2; ++m) {
#pragma unroll
      for (int r = 0; r < 8; ++r) {
        const float iv = 1.f / (1.f + expf(-acc[0][m][r]));
        const float fv = 1.f / (1.f + expf(-acc[1][m][r]));
        const float gv = tanhf(acc[2][m][r]);
        const float ov = 1.f / (1.f + expf(-acc[3][m][r]));
        const float cv = fv * cst[m][r] + iv * gv;
        cst[m][r] = cv;
        const float hv = ov * tanhf(cv);

        const int b = m * 16 + r + 8 * hs;
        const int n = w * 16 + lr;
        hbuf[b][n] = (half_t)hv;

        const size_t oidx = ((size_t)b * Tt + tt) * (2 * Ht) + dir * Ht + n;
        if (seq_f32) seq_f32[oidx] = hv;
        if (seq_f16) seq_f16[oidx] = (half_t)hv;
        if (t == Tt - 1) {
          const size_t sidx = (size_t)b * (2 * Ht) + dir * Ht + n;
          hT_out[sidx] = hv;
          cT_out[sidx] = cv;
        }
      }
    }
    __syncthreads(); // h_new fully published before next step reads
  }
}

// ---------------------------------------------------------------------------
extern "C" void kernel_launch(void* const* d_in, const int* in_sizes, int n_in,
                              void* d_out, int out_size, void* d_ws,
                              size_t ws_size, hipStream_t stream) {
  const float* x     = (const float*)d_in[0];
  const float* Wx_f0 = (const float*)d_in[1];
  const float* Wh_f0 = (const float*)d_in[2];
  const float* b_f0  = (const float*)d_in[3];
  const float* Wx_b0 = (const float*)d_in[4];
  const float* Wh_b0 = (const float*)d_in[5];
  const float* b_b0  = (const float*)d_in[6];
  const float* Wx_f1 = (const float*)d_in[7];
  const float* Wh_f1 = (const float*)d_in[8];
  const float* b_f1  = (const float*)d_in[9];
  const float* Wx_b1 = (const float*)d_in[10];
  const float* Wh_b1 = (const float*)d_in[11];
  const float* b_b1  = (const float*)d_in[12];
  float* out = (float*)d_out;

  // ---- workspace carve-up (all 256B aligned) ----
  char* ws = (char*)d_ws;
  size_t off = 0;
  auto carve = [&](size_t bytes) -> void* {
    void* p = ws + off;
    off += (bytes + 255) & ~(size_t)255;
    return p;
  };
  half_t* x_h     = (half_t*)carve((size_t)Bt * Tt * Et * 2);      // 16 MB
  half_t* WxT_f0h = (half_t*)carve((size_t)G4H * Et * 2);
  half_t* WxT_b0h = (half_t*)carve((size_t)G4H * Et * 2);
  half_t* WhT_f0h = (half_t*)carve((size_t)G4H * Ht * 2);
  half_t* WhT_b0h = (half_t*)carve((size_t)G4H * Ht * 2);
  half_t* WxT_f1h = (half_t*)carve((size_t)G4H * 2 * Ht * 2);
  half_t* WxT_b1h = (half_t*)carve((size_t)G4H * 2 * Ht * 2);
  half_t* WhT_f1h = (half_t*)carve((size_t)G4H * Ht * 2);
  half_t* WhT_b1h = (half_t*)carve((size_t)G4H * Ht * 2);
  half_t* x1_h    = (half_t*)carve((size_t)Bt * Tt * 2 * Ht * 2);  // 33 MB
  float*  Zf      = (float*)carve((size_t)Bt * Tt * G4H * 4);      // 134 MB
  float*  Zb      = (float*)carve((size_t)Bt * Tt * G4H * 4);      // 134 MB

  // ---- d_out regions: out [B,T,2H], h0, c0, h1, c1 ----
  const size_t SEQ = (size_t)Bt * Tt * 2 * Ht;
  const size_t ST  = (size_t)Bt * 2 * Ht;
  float* out_seq = out;
  float* h0 = out + SEQ;
  float* c0 = h0 + ST;
  float* h1 = c0 + ST;
  float* c1 = h1 + ST;

  // ---- 1. convert x to f16 ----
  {
    int n = Bt * Tt * Et;
    f32_to_f16_kernel<<<(n + 255) / 256, 256, 0, stream>>>(x, x_h, n);
  }
  // ---- 2. transpose+convert the 8 weight matrices ----
  auto tr = [&](const float* in, half_t* o, int K, int N) {
    int n = K * N;
    transpose_f16_kernel<<<(n + 255) / 256, 256, 0, stream>>>(in, o, K, N);
  };
  tr(Wx_f0, WxT_f0h, Et, G4H);
  tr(Wx_b0, WxT_b0h, Et, G4H);
  tr(Wh_f0, WhT_f0h, Ht, G4H);
  tr(Wh_b0, WhT_b0h, Ht, G4H);
  tr(Wx_f1, WxT_f1h, 2 * Ht, G4H);
  tr(Wx_b1, WxT_b1h, 2 * Ht, G4H);
  tr(Wh_f1, WhT_f1h, Ht, G4H);
  tr(Wh_b1, WhT_b1h, Ht, G4H);

  const int M = Bt * Tt;                                  // 16384 rows
  const int gemm_blocks = (M / 16) * (G4H / 32) / 8;      // 8192

  // ---- 3. layer-0 input projections ----
  gemm_xz_kernel<<<gemm_blocks, 256, 0, stream>>>(x_h, WxT_f0h, b_f0, Zf, M, Et);
  gemm_xz_kernel<<<gemm_blocks, 256, 0, stream>>>(x_h, WxT_b0h, b_b0, Zb, M, Et);

  // ---- 4. layer-0 recurrence (both dirs concurrently) ----
  lstm_rec_kernel<<<2, 1024, 0, stream>>>(Zf, Zb, WhT_f0h, WhT_b0h,
                                          (float*)nullptr, x1_h, h0, c0);

  // ---- 5. layer-1 input projections (K = 2H) ----
  gemm_xz_kernel<<<gemm_blocks, 256, 0, stream>>>(x1_h, WxT_f1h, b_f1, Zf, M, 2 * Ht);
  gemm_xz_kernel<<<gemm_blocks, 256, 0, stream>>>(x1_h, WxT_b1h, b_b1, Zb, M, 2 * Ht);

  // ---- 6. layer-1 recurrence -> final outputs ----
  lstm_rec_kernel<<<2, 1024, 0, stream>>>(Zf, Zb, WhT_f1h, WhT_b1h,
                                          out_seq, (half_t*)nullptr, h1, c1);
}
